// LinearImageAttention_12249246728268
// MI455X (gfx1250) — matrix operand order repaired and checked by reading, not compiled, not run on hardware
//
#include <hip/hip_runtime.h>
#include <hip/hip_bf16.h>

#define N_PIX 16384
#define CHN   256

typedef __attribute__((ext_vector_type(16))) __bf16 v16bf16;
typedef __attribute__((ext_vector_type(8)))  float  v8f;
typedef __attribute__((ext_vector_type(4)))  int    v4i;

union FragU { uint4 q[2]; v16bf16 v; };

#if __has_builtin(__builtin_amdgcn_global_load_async_to_lds_b128)
#define HAVE_ASYNC_LDS 1
#define TO_GLOBAL(p) ((__attribute__((address_space(1))) v4i*)(p))
#define TO_LDS(p)    ((__attribute__((address_space(3))) v4i*)(p))
#else
#define HAVE_ASYNC_LDS 0
#endif

__device__ __forceinline__ unsigned short f32_to_bf16(float f) {
    unsigned int u = __float_as_uint(f);
    unsigned int r = 0x7FFFu + ((u >> 16) & 1u);
    return (unsigned short)((u + r) >> 16);
}
__device__ __forceinline__ float bf16_to_f32(unsigned short h) {
    return __uint_as_float(((unsigned int)h) << 16);
}

// One wave computes a 16x16 fp32 tile: acc += A(16 x kLen) * B(kLen x 16).
// A in LDS row-major (lda elems/row), B in LDS column-major (ldb elems/col).
// lda/ldb and tile base offsets must be multiples of 8 elems (16B DS loads).
__device__ __forceinline__ v8f wmma_bf16_16x16(const unsigned short* As, int lda,
                                               const unsigned short* Bs, int ldb,
                                               int kLen, v8f acc, int lane)
{
    const int m  = lane & 15;
    const int kA = (lane & 16) ? 8  : 0;   // ISA A-frag lane split
    const int kB = (lane & 16) ? 8  : 0;   // ISA B-frag lane split
    for (int k0 = 0; k0 < kLen; k0 += 32) {
        FragU a, b;
        a.q[0] = *(const uint4*)(As + (size_t)m * lda + k0 + kA);
        a.q[1] = *(const uint4*)(As + (size_t)m * lda + k0 + kA + 16);
        b.q[0] = *(const uint4*)(Bs + (size_t)m * ldb + k0 + kB);
        b.q[1] = *(const uint4*)(Bs + (size_t)m * ldb + k0 + kB + 16);
        acc = __builtin_amdgcn_wmma_f32_16x16x32_bf16(false, a.v, false, b.v,
                                                      (short)0, acc, false, false);
    }
    return acc;
}

// ---------------------------------------------------------------------------
// K0: one-time weight prep.  Wcat = [Wq;Wk;Wv] in bf16 (640x256), fused bias,
// and Wout in bf16 (256x512).  Removes per-element source selection and all
// fp32->bf16 weight conversion from the hot GEMM kernels.
// ---------------------------------------------------------------------------
__global__ __launch_bounds__(256) void prep_kernel(
    const float* __restrict__ Wq, const float* __restrict__ bq,
    const float* __restrict__ Wk, const float* __restrict__ bk,
    const float* __restrict__ Wv, const float* __restrict__ bv,
    const float* __restrict__ Wout,
    unsigned short* __restrict__ Wcat, float* __restrict__ bcat,
    unsigned short* __restrict__ Woutb)
{
    const int tid = blockIdx.x * 256 + threadIdx.x;
    const int nt  = gridDim.x * 256;
    for (int i = tid; i < 640 * CHN; i += nt) {
        int r = i >> 8, c = i & 255;
        float v = (r < 512) ? Wq[(size_t)r * CHN + c]
                : (r < 576) ? Wk[(size_t)(r - 512) * CHN + c]
                            : Wv[(size_t)(r - 576) * CHN + c];
        Wcat[i] = f32_to_bf16(v);
    }
    for (int i = tid; i < 640; i += nt)
        bcat[i] = (i < 512) ? bq[i] : (i < 576) ? bk[i - 512] : bv[i - 576];
    for (int i = tid; i < 256 * 512; i += nt)
        Woutb[i] = f32_to_bf16(Wout[i]);
}

// ---------------------------------------------------------------------------
// K1: fused QKV projection.  Rows 0..511 = q, 512..575 = k, 576..639 = v.
// Per block: one batch, 64-pixel tile.  B (x tile, bf16, column-major) staged
// once; A (Wcat) staged in 128-row x 64-K chunks with vector b128 copies.
// Output routing is wave-uniform (16-row tiles never straddle a boundary).
// ---------------------------------------------------------------------------
__global__ __launch_bounds__(256) void qkv_kernel(
    const float* __restrict__ x,
    const unsigned short* __restrict__ Wcat, const float* __restrict__ bcat,
    unsigned short* __restrict__ qb,   // [8][512][16384] bf16
    float*          __restrict__ kf,   // [8][64][16384]  f32 (pre-softmax)
    unsigned short* __restrict__ vb)   // [8][64][16384]  bf16
{
    __shared__ unsigned short Bs[64][264];   // x tile  [pixel][K]   33.0 KB
    __shared__ unsigned short As[128][72];   // W chunk [row][K]     18.0 KB

    const int b    = blockIdx.y;
    const int pix0 = blockIdx.x * 64;
    const int tid  = threadIdx.x, lane = tid & 31, wave = tid >> 5;
    const float* xb = x + (size_t)b * CHN * N_PIX + pix0;

    for (int i = tid; i < 64 * CHN; i += 256) {      // transpose-stage x tile
        int p = i & 63, c = i >> 6;
        Bs[p][c] = f32_to_bf16(xb[(size_t)c * N_PIX + p]);
    }
    if (pix0 + 64 < N_PIX && tid < CHN)              // prefetch next x tile
        __builtin_prefetch(xb + (size_t)tid * N_PIX + 64, 0, 0);

    for (int rIter = 0; rIter < 5; ++rIter) {        // 5 * 128 = 640 rows
        v8f acc[4] = {};
        for (int kc = 0; kc < 4; ++kc) {             // 4 * 64 = 256 K
            __syncthreads();
            for (int i = tid; i < 128 * 8; i += 256) {   // b128 A-chunk copy
                int rr = i >> 3, sg = i & 7;
                *(uint4*)&As[rr][sg * 8] =
                    *(const uint4*)(Wcat + (size_t)(rIter * 128 + rr) * CHN +
                                    kc * 64 + sg * 8);
            }
            __syncthreads();
            for (int ct = 0; ct < 4; ++ct)
                acc[ct] = wmma_bf16_16x16(&As[wave * 16][0], 72,
                                          &Bs[ct * 16][kc * 64], 264,
                                          64, acc[ct], lane);
        }
        const int row0 = rIter * 128 + wave * 16;    // tile fully in q, k or v
        const int roff = ((lane >> 4) << 3);
        if (row0 < 512) {
            for (int ct = 0; ct < 4; ++ct) {
                int p = pix0 + ct * 16 + (lane & 15);
                for (int j = 0; j < 8; ++j) {
                    int r = row0 + roff + j;
                    qb[((size_t)b * 512 + r) * N_PIX + p] =
                        f32_to_bf16(acc[ct][j] + bcat[r]);
                }
            }
        } else if (row0 < 576) {
            for (int ct = 0; ct < 4; ++ct) {
                int p = pix0 + ct * 16 + (lane & 15);
                for (int j = 0; j < 8; ++j) {
                    int r = row0 + roff + j;
                    kf[((size_t)b * 64 + (r - 512)) * N_PIX + p] =
                        acc[ct][j] + bcat[r];
                }
            }
        } else {
            for (int ct = 0; ct < 4; ++ct) {
                int p = pix0 + ct * 16 + (lane & 15);
                for (int j = 0; j < 8; ++j) {
                    int r = row0 + roff + j;
                    vb[((size_t)b * 64 + (r - 576)) * N_PIX + p] =
                        f32_to_bf16(acc[ct][j] + bcat[r]);
                }
            }
        }
    }
}

// ---------------------------------------------------------------------------
// K2: softmax of k over the spatial axis (16384 elems per (b,d) row).
// ---------------------------------------------------------------------------
__global__ __launch_bounds__(256) void ksoftmax_kernel(
    const float* __restrict__ kf, unsigned short* __restrict__ kb)
{
    __shared__ float red[256];
    const int row = blockIdx.x;                   // b*64 + d
    const float* src = kf + (size_t)row * N_PIX;

    float mx = -3.4e38f;
    for (int i = threadIdx.x; i < N_PIX; i += 256) mx = fmaxf(mx, src[i]);
    red[threadIdx.x] = mx; __syncthreads();
    for (int s = 128; s > 0; s >>= 1) {
        if (threadIdx.x < s) red[threadIdx.x] = fmaxf(red[threadIdx.x], red[threadIdx.x + s]);
        __syncthreads();
    }
    mx = red[0]; __syncthreads();

    float sum = 0.f;
    for (int i = threadIdx.x; i < N_PIX; i += 256) sum += __expf(src[i] - mx);
    red[threadIdx.x] = sum; __syncthreads();
    for (int s = 128; s > 0; s >>= 1) {
        if (threadIdx.x < s) red[threadIdx.x] += red[threadIdx.x + s];
        __syncthreads();
    }
    const float inv = 1.f / red[0];

    unsigned short* dst = kb + (size_t)row * N_PIX;
    for (int i = threadIdx.x; i < N_PIX; i += 256)
        dst[i] = f32_to_bf16(__expf(src[i] - mx) * inv);
}

// ---------------------------------------------------------------------------
// K3: context[b] = k_sm(64 x n) * v^T(n x 64); writes context^T in bf16.
// Streaming reduction over n: double-buffered LDS chunks filled with
// GLOBAL_LOAD_ASYNC_TO_LDS_B128 (ASYNCcnt-tracked, next chunk in flight while
// current chunk is consumed by WMMA).  Falls back to plain copies if the
// async builtin is unavailable.  One block per batch.
// ---------------------------------------------------------------------------
__global__ __launch_bounds__(256) void context_kernel(
    const unsigned short* __restrict__ kb,
    const unsigned short* __restrict__ vb,
    unsigned short* __restrict__ ctxT)
{
    __shared__ unsigned short Ash[2][64][72];   // k_sm chunk [d][n]  2x9 KB
    __shared__ unsigned short Bsh[2][64][72];   // v    chunk [e][n]  2x9 KB
    const int b = blockIdx.x;
    const int tid = threadIdx.x, lane = tid & 31, wave = tid >> 5;
    const unsigned short* ks = kb + (size_t)b * 64 * N_PIX;
    const unsigned short* vs = vb + (size_t)b * 64 * N_PIX;
    const int nChunks = N_PIX / 64;             // 256 chunks of K=64

    auto stage = [&](int n0, int sel) {
        for (int i = tid; i < 64 * 8; i += 256) {      // 16B segments
            int r = i >> 3, sg = i & 7;
            const unsigned short* gk = ks + (size_t)r * N_PIX + n0 + sg * 8;
            const unsigned short* gv = vs + (size_t)r * N_PIX + n0 + sg * 8;
#if HAVE_ASYNC_LDS
            __builtin_amdgcn_global_load_async_to_lds_b128(
                TO_GLOBAL(gk), TO_LDS(&Ash[sel][r][sg * 8]), 0, 0);
            __builtin_amdgcn_global_load_async_to_lds_b128(
                TO_GLOBAL(gv), TO_LDS(&Bsh[sel][r][sg * 8]), 0, 0);
#else
            *(uint4*)&Ash[sel][r][sg * 8] = *(const uint4*)gk;
            *(uint4*)&Bsh[sel][r][sg * 8] = *(const uint4*)gv;
#endif
        }
    };

    v8f acc[2] = {};                            // 8 waves x 2 tiles = 4x4 tiles
    const int t0 = wave * 2;
    stage(0, 0);
    for (int c = 0; c < nChunks; ++c) {
        const int cur = c & 1;
        if (c + 1 < nChunks) stage((c + 1) * 64, cur ^ 1);
#if HAVE_ASYNC_LDS
        if (c + 1 < nChunks)
            asm volatile("s_wait_asynccnt 0x4" ::: "memory");  // prev chunk done
        else
            asm volatile("s_wait_asynccnt 0x0" ::: "memory");
#endif
        __syncthreads();
        for (int t = 0; t < 2; ++t) {
            int rt = (t0 + t) >> 2, ct = (t0 + t) & 3;
            acc[t] = wmma_bf16_16x16(&Ash[cur][rt * 16][0], 72,
                                     &Bsh[cur][ct * 16][0], 72, 64, acc[t], lane);
        }
        __syncthreads();
    }
    for (int t = 0; t < 2; ++t) {
        int rt = (t0 + t) >> 2, ct = (t0 + t) & 3;
        int e  = ct * 16 + (lane & 15);
        for (int j = 0; j < 8; ++j) {
            int d = rt * 16 + j + ((lane >> 4) << 3);
            ctxT[(size_t)b * 4096 + (size_t)e * 64 + d] = f32_to_bf16(acc[t][j]);
        }
    }
}

// ---------------------------------------------------------------------------
// K4: per-pixel softmax of q over d (within a head) fused with
// out[b][h*64+e][n] = sum_d ctxT[b][e][d] * softmax_q[b][h*64+d][n].
// One block per (128-pixel tile, head, batch).
// ---------------------------------------------------------------------------
__global__ __launch_bounds__(256) void attend_kernel(
    const unsigned short* __restrict__ qb,
    const unsigned short* __restrict__ ctxT,
    unsigned short* __restrict__ ob)
{
    __shared__ float          Qf [128][66];   // q tile fp32 [pixel][d]  33.0 KB
    __shared__ unsigned short Bsh[128][72];   // softmaxed q [pixel][d]  18.0 KB
    __shared__ unsigned short Ash[64][72];    // ctx^T       [e][d]       9.0 KB

    const int pix0 = blockIdx.x * 128, h = blockIdx.y, b = blockIdx.z;
    const int tid = threadIdx.x, lane = tid & 31, wave = tid >> 5;
    const unsigned short* qsrc = qb + ((size_t)b * 512 + h * 64) * N_PIX + pix0;

    for (int i = tid; i < 64 * 128; i += 256) {
        int d = i >> 7, p = i & 127;
        Qf[p][d] = bf16_to_f32(qsrc[(size_t)d * N_PIX + p]);
    }
    for (int i = tid; i < 64 * 8; i += 256) {       // b128 copy of ctx^T
        int e = i >> 3, sg = i & 7;
        *(uint4*)&Ash[e][sg * 8] =
            *(const uint4*)(ctxT + (size_t)b * 4096 + (size_t)e * 64 + sg * 8);
    }
    __syncthreads();

    if (tid < 128) {                          // softmax over d for column tid
        float mx = -3.4e38f;
        for (int d = 0; d < 64; ++d) mx = fmaxf(mx, Qf[tid][d]);
        float sum = 0.f;
        for (int d = 0; d < 64; ++d) { float e = __expf(Qf[tid][d] - mx); Qf[tid][d] = e; sum += e; }
        float inv = 1.f / sum;
        for (int d = 0; d < 64; ++d) Bsh[tid][d] = f32_to_bf16(Qf[tid][d] * inv);
    }
    __syncthreads();

    v8f acc[4] = {};                          // 8 waves x 4 tiles = 4x8 tiles
    for (int t = 0; t < 4; ++t) {
        int tile = wave * 4 + t, rt = tile >> 3, ct = tile & 7;
        acc[t] = wmma_bf16_16x16(&Ash[rt * 16][0], 72,
                                 &Bsh[ct * 16][0], 72, 64, acc[t], lane);
    }
    for (int t = 0; t < 4; ++t) {
        int tile = wave * 4 + t, rt = tile >> 3, ct = tile & 7;
        int p = pix0 + ct * 16 + (lane & 15);
        for (int j = 0; j < 8; ++j) {
            int e = rt * 16 + j + ((lane >> 4) << 3);
            ob[((size_t)b * 512 + h * 64 + e) * N_PIX + p] = f32_to_bf16(acc[t][j]);
        }
    }
}

// ---------------------------------------------------------------------------
// K5: output projection  out = Wout(256x512) * ob(512 x n) + bout, fp32 out.
// One block per (64-pixel tile, batch); A (pre-converted bf16 Wout) and B
// staged in 64-K chunks.
// ---------------------------------------------------------------------------
__global__ __launch_bounds__(256) void outproj_kernel(
    const unsigned short* __restrict__ ob,
    const unsigned short* __restrict__ Woutb, const float* __restrict__ bout,
    float* __restrict__ out)
{
    __shared__ unsigned short Ash[128][72];   // Wout chunk [row][K]
    __shared__ unsigned short Bsh[64][72];    // ob   chunk [pixel][K]
    const int pix0 = blockIdx.x * 64, b = blockIdx.y;
    const int tid = threadIdx.x, lane = tid & 31, wave = tid >> 5;
    const unsigned short* osrc = ob + (size_t)b * 512 * N_PIX + pix0;

    for (int rIter = 0; rIter < 2; ++rIter) {        // 2 * 128 = 256 rows
        v8f acc[4] = {};
        for (int kc = 0; kc < 8; ++kc) {             // 8 * 64 = 512 K
            __syncthreads();
            for (int i = tid; i < 128 * 8; i += 256) {   // b128 A-chunk copy
                int rr = i >> 3, sg = i & 7;
                *(uint4*)&Ash[rr][sg * 8] =
                    *(const uint4*)(Woutb + (size_t)(rIter * 128 + rr) * 512 +
                                    kc * 64 + sg * 8);
            }
            for (int i = tid; i < 64 * 64; i += 256) {   // transpose B-chunk
                int p = i >> 6, k = i & 63;
                Bsh[p][k] = osrc[(size_t)(kc * 64 + k) * N_PIX + p];
            }
            __syncthreads();
            for (int ct = 0; ct < 4; ++ct)
                acc[ct] = wmma_bf16_16x16(&Ash[wave * 16][0], 72,
                                          &Bsh[ct * 16][0], 72, 64, acc[ct], lane);
        }
        const int row0 = rIter * 128 + wave * 16;
        for (int ct = 0; ct < 4; ++ct) {
            int p = pix0 + ct * 16 + (lane & 15);
            for (int j = 0; j < 8; ++j) {
                int r = row0 + j + ((lane >> 4) << 3);
                out[((size_t)b * 256 + r) * N_PIX + p] = acc[ct][j] + bout[r];
            }
        }
    }
}

// ---------------------------------------------------------------------------
extern "C" void kernel_launch(void* const* d_in, const int* in_sizes, int n_in,
                              void* d_out, int out_size, void* d_ws, size_t ws_size,
                              hipStream_t stream)
{
    const float* x    = (const float*)d_in[0];
    const float* Wq   = (const float*)d_in[1];
    const float* bq   = (const float*)d_in[2];
    const float* Wk   = (const float*)d_in[3];
    const float* bk   = (const float*)d_in[4];
    const float* Wv   = (const float*)d_in[5];
    const float* bv   = (const float*)d_in[6];
    const float* Wout = (const float*)d_in[7];
    const float* bout = (const float*)d_in[8];

    char* ws = (char*)d_ws;
    unsigned short* qb    = (unsigned short*)ws; ws += (size_t)8 * 512 * N_PIX * 2;
    float*          kf    = (float*)ws;          ws += (size_t)8 * 64  * N_PIX * 4;
    unsigned short* vb    = (unsigned short*)ws; ws += (size_t)8 * 64  * N_PIX * 2;
    unsigned short* kb    = (unsigned short*)ws; ws += (size_t)8 * 64  * N_PIX * 2;
    unsigned short* ctxT  = (unsigned short*)ws; ws += (size_t)8 * 64  * 64    * 2;
    unsigned short* obuf  = (unsigned short*)ws; ws += (size_t)8 * 512 * N_PIX * 2;
    unsigned short* Wcat  = (unsigned short*)ws; ws += (size_t)640 * CHN * 2;
    float*          bcat  = (float*)ws;          ws += (size_t)640 * 4;
    unsigned short* Woutb = (unsigned short*)ws;

    prep_kernel    <<<dim3(128),               256, 0, stream>>>(Wq, bq, Wk, bk, Wv, bv, Wout, Wcat, bcat, Woutb);
    qkv_kernel     <<<dim3(N_PIX / 64, 8),     256, 0, stream>>>(x, Wcat, bcat, qb, kf, vb);
    ksoftmax_kernel<<<dim3(8 * 64),            256, 0, stream>>>(kf, kb);
    context_kernel <<<dim3(8),                 256, 0, stream>>>(kb, vb, ctxT);
    attend_kernel  <<<dim3(N_PIX / 128, 8, 8), 256, 0, stream>>>(qb, ctxT, obuf);
    outproj_kernel <<<dim3(N_PIX / 64, 8),     256, 0, stream>>>(obuf, Woutb, bout, (float*)d_out);
}
